// Classifier_79370995630553
// MI455X (gfx1250) — matrix operand order repaired and checked by reading, not compiled, not run on hardware
//
#include <hip/hip_runtime.h>
#include <hip/hip_bf16.h>

// ---------------------------------------------------------------------------
// GRU classifier for MI455X (gfx1250, wave32, WMMA 16x16x32 bf16)
//   out = FC( GRU(x) )      x:[256,512,512] f32 -> out:[256,101] f32
//
// K1: xg = x @ W_ih^T + b_ih   (131072x768x512 bf16 WMMA GEMM, ~103 GFLOP)
// K2: persistent-RNN recurrence: W_hh register-stationary per wave; gate
//     triples (r,z,n) of a hidden unit land in the same lane/acc-slot, so
//     the nonlinearity fuses in registers (no LDS gate buffer).
// K3: WMMA FC head.
// ---------------------------------------------------------------------------

typedef __bf16 bf16_t;
typedef __attribute__((ext_vector_type(16))) __bf16 v16bf;
typedef __attribute__((ext_vector_type(8)))  __bf16 v8bf;
typedef __attribute__((ext_vector_type(8)))  float  v8f;

#define DINLINE __device__ __forceinline__

constexpr int BB  = 256;   // batch
constexpr int TT  = 512;   // timesteps
constexpr int NIN = 512;   // input features
constexpr int HH  = 256;   // hidden
constexpr int GG  = 768;   // 3*H (gates r,z,n)
constexpr int NC  = 101;   // classes
constexpr int NCP = 112;   // classes padded to 16
constexpr long long MT = (long long)BB * TT;  // 131072 rows of xg

DINLINE bf16_t f2bf(float f) { return (bf16_t)f; }   // native cvt, RNE

DINLINE v8f vzero8() { v8f z = {0.f,0.f,0.f,0.f,0.f,0.f,0.f,0.f}; return z; }

// B-fragment: 16 contiguous bf16 (32B, 32B-aligned in all our uses)
DINLINE v16bf load16bf(const bf16_t* p) { return *(const v16bf*)p; }

// A-fragment: two contiguous 8-element runs (contiguous K order, matching B:
// lane half=0 -> K = {0..15}; half=1 -> K = {16..31})
DINLINE v16bf joinA(const bf16_t* lo, const bf16_t* hi) {
  v8bf a = *(const v8bf*)lo;
  v8bf b = *(const v8bf*)hi;
  return __builtin_shufflevector(a, b, 0,1,2,3,4,5,6,7,8,9,10,11,12,13,14,15);
}

DINLINE v8bf conv8(const float* p) {  // 8 f32 -> 8 bf16
  float4 a = *(const float4*)p;
  float4 b = *(const float4*)(p + 4);
  v8bf r;
  r[0]=f2bf(a.x); r[1]=f2bf(a.y); r[2]=f2bf(a.z); r[3]=f2bf(a.w);
  r[4]=f2bf(b.x); r[5]=f2bf(b.y); r[6]=f2bf(b.z); r[7]=f2bf(b.w);
  return r;
}

DINLINE v16bf join8(v8bf a, v8bf b) {
  return __builtin_shufflevector(a, b, 0,1,2,3,4,5,6,7,8,9,10,11,12,13,14,15);
}

DINLINE v8f wmma_bf16(v16bf a, v16bf b, v8f c) {
  // (neg_a, A, neg_b, B, c_mod, C, reuse_a, reuse_b)
  return __builtin_amdgcn_wmma_f32_16x16x32_bf16(false, a, false, b, (short)0, c,
                                                 false, false);
}

DINLINE float sigmoidf_(float x) { return 1.f / (1.f + __expf(-x)); }
DINLINE float tanhf_(float x) {
  float e = __expf(-2.f * x);
  return (1.f - e) / (1.f + e);
}

// ---------------------------------------------------------------------------
// K0: convert weights to bf16 (fc_w zero-padded to [112,256])
// ---------------------------------------------------------------------------
__global__ __launch_bounds__(256) void k_convert(
    const float* __restrict__ Wih, const float* __restrict__ Whh,
    const float* __restrict__ fcw,
    bf16_t* __restrict__ wih_b, bf16_t* __restrict__ whh_b,
    bf16_t* __restrict__ fcw_b) {
  int i = blockIdx.x * 256 + threadIdx.x;
  if (i < GG * NIN) wih_b[i] = f2bf(Wih[i]);
  if (i < GG * HH)  whh_b[i] = f2bf(Whh[i]);
  if (i < NCP * HH) {
    int r = i >> 8, c = i & 255;
    fcw_b[i] = f2bf(r < NC ? fcw[r * HH + c] : 0.f);
  }
}

// ---------------------------------------------------------------------------
// K1: xg = x @ W_ih^T + b_ih   (M=131072, N=768, K=512), bf16 out
// Block = 256 thr = 8 waves as 2(M) x 4(N); block tile 32x768 -> x read once.
// Wave tile: 16(M) x 192(N) = 12 accs. B loads batched 4 tiles at a time so
// the WMMAs behind them wait on partial load counts instead of loadcnt==0.
// ---------------------------------------------------------------------------
__global__ __launch_bounds__(256) void k_xgemm(
    const float* __restrict__ x, const bf16_t* __restrict__ wih,
    const float* __restrict__ bih, bf16_t* __restrict__ xg) {
  const int lane = threadIdx.x & 31;
  const int wave = threadIdx.x >> 5;
  const int half = lane >> 4;
  const int l16  = lane & 15;
  const long long m0 = (long long)blockIdx.x * 32 + (wave >> 2) * 16;
  const int nbase = (wave & 3) * 192;

  v8f acc[12];
  #pragma unroll
  for (int i = 0; i < 12; ++i) acc[i] = vzero8();

  const float*  xrow  = x + (m0 + l16) * NIN;
  const bf16_t* wbase = wih + (size_t)(nbase + l16) * NIN + half * 16;

  #pragma unroll 2
  for (int kk = 0; kk < NIN; kk += 32) {
    v16bf va = join8(conv8(xrow + kk + half * 16),
                     conv8(xrow + kk + 8 + half * 16));
    #pragma unroll
    for (int grp = 0; grp < 3; ++grp) {
      v16bf vb[4];
      #pragma unroll
      for (int q = 0; q < 4; ++q)
        vb[q] = load16bf(wbase + (size_t)(grp * 4 + q) * 16 * NIN + kk);
      #pragma unroll
      for (int q = 0; q < 4; ++q)
        acc[grp * 4 + q] = wmma_bf16(va, vb[q], acc[grp * 4 + q]);
    }
  }

  #pragma unroll
  for (int tI = 0; tI < 12; ++tI) {
    const int n = nbase + tI * 16 + l16;
    const float bias = bih[n];
    #pragma unroll
    for (int r = 0; r < 8; ++r) {
      const long long gm = m0 + r + 8 * half;
      xg[gm * GG + n] = f2bf(acc[tI][r] + bias);
    }
  }
}

// ---------------------------------------------------------------------------
// K2: GRU recurrence, persistent-RNN style. 16 blocks x 16 batch rows.
//  - Wave w owns hidden units [w*32, w*32+32): 2 unit-groups x 3 gates
//    = 6 WMMA N-tiles. Its 48 W_hh B-fragments (384 VGPRs) are loaded ONCE
//    and stay register-stationary for all 512 steps.
//  - D-tile layout: lane l, acc slot r <-> (unit = base+l%16, row = r+8*(l/16)),
//    identical for the r/z/n tiles -> gate math fuses entirely in registers.
//  - h slice (16x256 bf16) lives in LDS; A-fragments re-read from it each step.
//  - xg_t streamed from HBM, next step prefetched (global_prefetch_b8).
// ---------------------------------------------------------------------------
__global__ __launch_bounds__(256) void k_gru(
    const bf16_t* __restrict__ xg, const bf16_t* __restrict__ whh,
    const float* __restrict__ bhh_g, bf16_t* __restrict__ hT) {
  __shared__ bf16_t h_lds[16][264];   // 16 rows x 256 (+8 pad), bf16

  const int tid  = threadIdx.x;
  const int lane = tid & 31;
  const int wave = tid >> 5;
  const int half = lane >> 4;
  const int l16  = lane & 15;
  const int wbase = wave * 32;               // first hidden unit of this wave
  const long long grow = (long long)blockIdx.x * 16;

  for (int i = tid; i < 16 * 264; i += 256) ((bf16_t*)h_lds)[i] = f2bf(0.f);

  // ---- preload register-stationary W_hh fragments: Bf[group][gate][kk8] ----
  v16bf Bf[2][3][8];
  #pragma unroll
  for (int g = 0; g < 2; ++g) {
    #pragma unroll
    for (int gate = 0; gate < 3; ++gate) {
      const bf16_t* wrow =
          whh + (size_t)(gate * HH + wbase + g * 16 + l16) * HH + half * 16;
      #pragma unroll
      for (int k8 = 0; k8 < 8; ++k8) Bf[g][gate][k8] = load16bf(wrow + k8 * 32);
    }
  }

  // per-lane hidden-side gate biases (constant over t)
  float bR[2], bZ[2], bN[2];
  #pragma unroll
  for (int g = 0; g < 2; ++g) {
    const int j = wbase + g * 16 + l16;
    bR[g] = bhh_g[j];
    bZ[g] = bhh_g[j + HH];
    bN[g] = bhh_g[j + 2 * HH];
  }

  __syncthreads();

  for (int t = 0; t < TT; ++t) {
    // prefetch next timestep's xg slice (16 rows x 1536B = 192 cachelines)
    if (t + 1 < TT && tid < 192) {
      const int pm = tid / 12, pc = (tid % 12) * 64;
      __builtin_prefetch(xg + ((grow + pm) * TT + (t + 1)) * GG + pc, 0, 1);
    }

    // ---- hg = h @ W_hh^T : 48 WMMAs, B operands already in registers ----
    v8f aR[2], aZ[2], aN[2];
    #pragma unroll
    for (int g = 0; g < 2; ++g) { aR[g] = vzero8(); aZ[g] = vzero8(); aN[g] = vzero8(); }

    #pragma unroll
    for (int k8 = 0; k8 < 8; ++k8) {
      const int kk = k8 * 32;
      v16bf va = joinA(&h_lds[l16][kk + half * 16],
                       &h_lds[l16][kk + 8 + half * 16]);
      #pragma unroll
      for (int g = 0; g < 2; ++g) {
        aR[g] = wmma_bf16(va, Bf[g][0][k8], aR[g]);
        aZ[g] = wmma_bf16(va, Bf[g][1][k8], aZ[g]);
        aN[g] = wmma_bf16(va, Bf[g][2][k8], aN[g]);
      }
    }
    __syncthreads();   // all A-fragment reads of h done

    // ---- fused gates + h update, all operands register/lane-local ----
    #pragma unroll
    for (int g = 0; g < 2; ++g) {
      const int j = wbase + g * 16 + l16;
      #pragma unroll
      for (int ri = 0; ri < 8; ++ri) {
        const int m = ri + 8 * half;
        const bf16_t* xp = xg + ((grow + m) * TT + t) * GG + j;
        const float xr = (float)xp[0];
        const float xz = (float)xp[HH];
        const float xn = (float)xp[2 * HH];
        const float r = sigmoidf_(xr + aR[g][ri] + bR[g]);
        const float z = sigmoidf_(xz + aZ[g][ri] + bZ[g]);
        const float nn = tanhf_(xn + r * (aN[g][ri] + bN[g]));
        const float hp = (float)h_lds[m][j];
        h_lds[m][j] = f2bf((1.f - z) * nn + z * hp);
      }
    }
    __syncthreads();   // h fully updated before next step's A reads
  }

  // ---- write final hidden state ----
  #pragma unroll
  for (int g = 0; g < 2; ++g) {
    const int j = wbase + g * 16 + l16;
    #pragma unroll
    for (int ri = 0; ri < 8; ++ri) {
      const int m = ri + 8 * half;
      hT[(grow + m) * HH + j] = h_lds[m][j];
    }
  }
}

// ---------------------------------------------------------------------------
// K3: out = hT @ fc_w^T + fc_b   (M=256, N=112 padded, K=256)
// 112 wave-tiles of 16x16 -> 14 blocks of 8 waves.
// ---------------------------------------------------------------------------
__global__ __launch_bounds__(256) void k_fc(
    const bf16_t* __restrict__ hT, const bf16_t* __restrict__ fcw,
    const float* __restrict__ fcb, float* __restrict__ out) {
  const int lane = threadIdx.x & 31;
  const int wave = threadIdx.x >> 5;
  const int half = lane >> 4;
  const int l16  = lane & 15;
  const int wl = blockIdx.x * 8 + wave;   // 0..111
  const int m0 = (wl / 7) * 16;
  const int n0 = (wl % 7) * 16;

  v8f acc = vzero8();
  const bf16_t* arow = hT + (size_t)(m0 + l16) * HH;
  #pragma unroll
  for (int kk = 0; kk < HH; kk += 32) {
    v16bf va = joinA(arow + kk + half * 16, arow + kk + 8 + half * 16);
    v16bf vb = load16bf(fcw + (size_t)(n0 + l16) * HH + kk + half * 16);
    acc = wmma_bf16(va, vb, acc);
  }

  const int n = n0 + l16;
  const float bias = (n < NC) ? fcb[n] : 0.f;
  #pragma unroll
  for (int r = 0; r < 8; ++r) {
    const int m = m0 + r + 8 * half;
    if (n < NC) out[m * NC + n] = acc[r] + bias;
  }
}

// ---------------------------------------------------------------------------
extern "C" void kernel_launch(void* const* d_in, const int* in_sizes, int n_in,
                              void* d_out, int out_size, void* d_ws,
                              size_t ws_size, hipStream_t stream) {
  const float* x   = (const float*)d_in[0];
  const float* Wih = (const float*)d_in[1];
  const float* Whh = (const float*)d_in[2];
  const float* bih = (const float*)d_in[3];
  const float* bhh = (const float*)d_in[4];
  const float* fcw = (const float*)d_in[5];
  const float* fcb = (const float*)d_in[6];
  float* out = (float*)d_out;

  // Workspace layout (bytes): xg bf16 | W_ih bf16 | W_hh bf16 | fc_w bf16 | hT
  char* ws = (char*)d_ws;
  size_t off = 0;
  bf16_t* xgb   = (bf16_t*)(ws + off); off += (size_t)MT * GG * 2;   // 192 MiB
  bf16_t* wih_b = (bf16_t*)(ws + off); off += (size_t)GG * NIN * 2;
  bf16_t* whh_b = (bf16_t*)(ws + off); off += (size_t)GG * HH * 2;
  bf16_t* fcw_b = (bf16_t*)(ws + off); off += (size_t)NCP * HH * 2;
  bf16_t* hTb   = (bf16_t*)(ws + off); off += (size_t)BB * HH * 2;
  (void)ws_size; (void)in_sizes; (void)n_in; (void)out_size;

  k_convert<<<(GG * NIN + 255) / 256, 256, 0, stream>>>(Wih, Whh, fcw,
                                                        wih_b, whh_b, fcw_b);
  k_xgemm<<<(int)(MT / 32), 256, 0, stream>>>(x, wih_b, bih, xgb);
  k_gru<<<BB / 16, 256, 0, stream>>>(xgb, whh_b, bhh, hTb);
  k_fc<<<14, 256, 0, stream>>>(hTb, fcw_b, fcb, out);
}
